// Net_51960514347273
// MI455X (gfx1250) — matrix operand order repaired and checked by reading, not compiled, not run on hardware
//
#include <hip/hip_runtime.h>
#include <hip/hip_bf16.h>

typedef __attribute__((ext_vector_type(16))) _Float16 v16h;
typedef __attribute__((ext_vector_type(8)))  float    v8f;

// ---------------------------------------------------------------------------
// Degree / normalization kernels
// ---------------------------------------------------------------------------
__global__ void deg_init(float* __restrict__ deg, int n) {
    int i = blockIdx.x * blockDim.x + threadIdx.x;
    if (i < n) deg[i] = 1.0f;              // self-loop contribution
}

__global__ void deg_count(const int* __restrict__ dst, float* __restrict__ deg, int e) {
    int i = blockIdx.x * blockDim.x + threadIdx.x;
    if (i < e) atomicAdd(&deg[dst[i]], 1.0f);
}

__global__ void deg_rsqrt(float* __restrict__ deg, int n) {
    int i = blockIdx.x * blockDim.x + threadIdx.x;
    if (i < n) deg[i] = rsqrtf(deg[i]);    // deg >= 1 always (self loop)
}

// ---------------------------------------------------------------------------
// WMMA GEMM:  Y[M,N_] = X[M,K_] * W[K_,N_]   (f32 in, f16 multiply, f32 accum)
// One wave computes a 16-row strip across all N_/16 column tiles.
// Block = 256 threads = 8 waves = 128 rows per block.
// * A strip preloaded with batched float4 loads, converted to v16h fragments.
// * W pre-packed in LDS in the exact B-fragment register layout; per k-tile
//   all NT fragments are fetched (2x ds_load_b128 each) before the WMMA chain.
// ---------------------------------------------------------------------------
template <int K_, int N_>
__global__ void gemm_wmma(const float* __restrict__ X, const float* __restrict__ Wg,
                          float* __restrict__ Y, int M) {
    constexpr int KT = K_ / 32;
    constexpr int NT = N_ / 16;

    // sB[(nt*KT + kt)*32 + lane] = the 16 halves lane needs for tile (nt,kt)
    __shared__ v16h sB[NT * KT * 32];
    {
        _Float16* sBh = (_Float16*)sB;
        for (int i = threadIdx.x; i < K_ * N_; i += blockDim.x) {
            const int k = i / N_;           // coalesced read of W in row-major order
            const int n = i - k * N_;
            // B 32x16 f16 layout: lane = half_id*16 + (n%16); half_id = (k%32/8)&1, j = k%8 + 8*(k%32/16)
            const int kt      = k >> 5;
            const int kr      = k & 31;
            const int half_id = (kr >> 3) & 1;
            const int j       = (kr & 7) + ((kr >> 4) << 3);
            const int nt      = n >> 4;
            const int lane    = half_id * 16 + (n & 15);
            sBh[(((nt * KT + kt) * 32) + lane) * 16 + j] = (_Float16)Wg[i];
        }
    }
    __syncthreads();

    const int wave    = threadIdx.x >> 5;
    const int lane    = threadIdx.x & 31;
    const int half_id = lane >> 4;          // 0: lanes 0-15, 1: lanes 16-31
    const int mr      = lane & 15;
    const int strips  = (M + 15) >> 4;
    const int strip   = blockIdx.x * (blockDim.x >> 5) + wave;
    if (strip >= strips) return;
    const int m0 = strip << 4;

    // ---- preload entire A strip row (K_ floats) as KT v16h fragments ----
    // 16-bit A 16x32 layout: lanes 0-15 hold K=[0..8)+[16..24),
    //                        lanes 16-31 hold K=[8..16)+[24..32)
    int arow = m0 + mr;
    if (arow >= M) arow = M - 1;            // clamped read for partial strips
    const float* __restrict__ xr = X + (size_t)arow * K_;

    v16h afrag[KT];
#pragma unroll
    for (int kt = 0; kt < KT; ++kt) {
        const int ka = kt * 32 + half_id * 8;
        const float4 x0 = *reinterpret_cast<const float4*>(xr + ka);
        const float4 x1 = *reinterpret_cast<const float4*>(xr + ka + 4);
        const float4 x2 = *reinterpret_cast<const float4*>(xr + ka + 16);
        const float4 x3 = *reinterpret_cast<const float4*>(xr + ka + 20);
        afrag[kt][0]  = (_Float16)x0.x; afrag[kt][1]  = (_Float16)x0.y;
        afrag[kt][2]  = (_Float16)x0.z; afrag[kt][3]  = (_Float16)x0.w;
        afrag[kt][4]  = (_Float16)x1.x; afrag[kt][5]  = (_Float16)x1.y;
        afrag[kt][6]  = (_Float16)x1.z; afrag[kt][7]  = (_Float16)x1.w;
        afrag[kt][8]  = (_Float16)x2.x; afrag[kt][9]  = (_Float16)x2.y;
        afrag[kt][10] = (_Float16)x2.z; afrag[kt][11] = (_Float16)x2.w;
        afrag[kt][12] = (_Float16)x3.x; afrag[kt][13] = (_Float16)x3.y;
        afrag[kt][14] = (_Float16)x3.z; afrag[kt][15] = (_Float16)x3.w;
    }

    v8f acc[NT];
#pragma unroll
    for (int t = 0; t < NT; ++t) acc[t] = (v8f){0.f,0.f,0.f,0.f,0.f,0.f,0.f,0.f};

#pragma unroll
    for (int kt = 0; kt < KT; ++kt) {
        v16h bf[NT];
#pragma unroll
        for (int nt = 0; nt < NT; ++nt)
            bf[nt] = sB[(nt * KT + kt) * 32 + lane];
#pragma unroll
        for (int nt = 0; nt < NT; ++nt)
            acc[nt] = __builtin_amdgcn_wmma_f32_16x16x32_f16(
                false, afrag[kt], false, bf[nt], (short)0, acc[nt], false, false);
    }

    // C layout: lanes 0-15: VGPR r -> (M=r, N=lane); lanes 16-31: (M=8+r, N=lane-16)
#pragma unroll
    for (int nt = 0; nt < NT; ++nt) {
#pragma unroll
        for (int r = 0; r < 8; ++r) {
            const int row = m0 + r + 8 * half_id;
            if (row < M) Y[(size_t)row * N_ + nt * 16 + mr] = acc[nt][r];
        }
    }
}

// ---------------------------------------------------------------------------
// Aggregation: out = D^-1/2 (A+I) D^-1/2 (XW) + b
// ---------------------------------------------------------------------------
__global__ void agg_init(float* __restrict__ acc, const float* __restrict__ xw,
                         const float* __restrict__ dinv, const float* __restrict__ bias,
                         int n, int F) {
    int i = blockIdx.x * blockDim.x + threadIdx.x;
    if (i < n * F) {
        int node = i / F, f = i - node * F;
        float dv = dinv[node];
        acc[i] = xw[i] * dv * dv + bias[f];     // self loop + bias
    }
}

// Each wave processes (128/F) edges so all 32 lanes stay busy:
//   F=128 -> 1 edge/wave (32 lanes x float4), F=64 -> 2 edges/wave (16 lanes each).
__global__ void edge_scatter(const int* __restrict__ src, const int* __restrict__ dst,
                             const float* __restrict__ dinv, const float* __restrict__ xw,
                             float* __restrict__ acc, int e, int F) {
    const int lanesPerEdge = F >> 2;                     // 32 or 16
    const int epw          = 32 / lanesPerEdge;          // 1 or 2
    const int w    = blockIdx.x * (blockDim.x >> 5) + (threadIdx.x >> 5);
    const int lane = threadIdx.x & 31;
    const int g    = lane / lanesPerEdge;                // edge slot within wave
    const int lf   = lane - g * lanesPerEdge;            // lane within edge
    const int ei   = w * epw + g;
    if (ei >= e) return;
    const int s = src[ei], d = dst[ei];
    const float c = dinv[s] * dinv[d];
    const int f = lf << 2;
    const float4 v = *reinterpret_cast<const float4*>(xw + (size_t)s * F + f);
    float* o = acc + (size_t)d * F + f;
    atomicAdd(o + 0, v.x * c);
    atomicAdd(o + 1, v.y * c);
    atomicAdd(o + 2, v.z * c);
    atomicAdd(o + 3, v.w * c);
}

__global__ void relu_k(float* __restrict__ p, int n) {
    int i = blockIdx.x * blockDim.x + threadIdx.x;
    if (i < n) p[i] = fmaxf(p[i], 0.0f);
}

// ---------------------------------------------------------------------------
// Decode: logits[e] = dot(z[a], z[b]) ; one wave per label edge
// ---------------------------------------------------------------------------
__global__ void decode_dot(const int* __restrict__ ea, const int* __restrict__ eb,
                           const float* __restrict__ Z, float* __restrict__ out,
                           int el, int F) {
    int w    = blockIdx.x * (blockDim.x >> 5) + (threadIdx.x >> 5);
    int lane = threadIdx.x & 31;
    if (w >= el) return;
    int a = ea[w], b = eb[w];
    float p = 0.f;
    for (int f = lane; f < F; f += 32)
        p += Z[(size_t)a * F + f] * Z[(size_t)b * F + f];
#pragma unroll
    for (int off = 16; off; off >>= 1) p += __shfl_xor(p, off, 32);
    if (lane == 0) out[w] = p;
}

// ---------------------------------------------------------------------------
extern "C" void kernel_launch(void* const* d_in, const int* in_sizes, int n_in,
                              void* d_out, int out_size, void* d_ws, size_t ws_size,
                              hipStream_t stream) {
    const float* x   = (const float*)d_in[0];
    const int*   ei  = (const int*)d_in[1];
    const int*   eli = (const int*)d_in[2];
    const float* W1  = (const float*)d_in[3];
    const float* b1  = (const float*)d_in[4];
    const float* W2  = (const float*)d_in[5];
    const float* b2  = (const float*)d_in[6];
    float*       out = (float*)d_out;

    const int HID = in_sizes[4];                // 128
    const int OUT = in_sizes[6];                // 64
    const int IN  = in_sizes[3] / HID;          // 128
    const int N   = in_sizes[0] / IN;           // 50000
    const int E   = in_sizes[1] / 2;            // 1.6M
    const int EL  = in_sizes[2] / 2;            // 200K

    // workspace layout (floats): dinv[N] | bufA[N*HID] | bufB[N*HID]
    float* dinv = (float*)d_ws;
    float* bufA = dinv + N;                     // xW1, then reused as xW2
    float* bufB = bufA + (size_t)N * HID;       // h,   then reused as z

    const int T = 256;
    auto cdiv = [](int a, int b) { return (a + b - 1) / b; };

    // --- normalization coefficients (shared by both layers) ---
    deg_init<<<cdiv(N, T), T, 0, stream>>>(dinv, N);
    deg_count<<<cdiv(E, T), T, 0, stream>>>(ei + E, dinv, E);
    deg_rsqrt<<<cdiv(N, T), T, 0, stream>>>(dinv, N);

    const int strips     = cdiv(N, 16);
    const int gemmBlocks = cdiv(strips, T / 32);

    // --- layer 1: h = relu(agg(x @ W1) + b1) ---
    gemm_wmma<128, 128><<<gemmBlocks, T, 0, stream>>>(x, W1, bufA, N);
    agg_init<<<cdiv(N * HID, T), T, 0, stream>>>(bufB, bufA, dinv, b1, N, HID);
    {
        const int epw = 128 / HID;              // edges per wave
        edge_scatter<<<cdiv(cdiv(E, epw), T / 32), T, 0, stream>>>(
            ei, ei + E, dinv, bufA, bufB, E, HID);
    }
    relu_k<<<cdiv(N * HID, T), T, 0, stream>>>(bufB, N * HID);

    // --- layer 2: z = agg(h @ W2) + b2   (bufA reused for xW2, z overwrites h) ---
    gemm_wmma<128, 64><<<gemmBlocks, T, 0, stream>>>(bufB, W2, bufA, N);
    agg_init<<<cdiv(N * OUT, T), T, 0, stream>>>(bufB, bufA, dinv, b2, N, OUT);
    {
        const int epw = 128 / OUT;              // 2 edges per wave for F=64
        edge_scatter<<<cdiv(cdiv(E, epw), T / 32), T, 0, stream>>>(
            ei, ei + E, dinv, bufA, bufB, E, OUT);
    }

    // --- decode ---
    decode_dot<<<cdiv(EL, T / 32), T, 0, stream>>>(eli, eli + EL, bufB, out, EL, OUT);
}
